// TopDownBilinearLSTMEncoder_75797582839904
// MI455X (gfx1250) — compile-verified, matrix operand order checked by reading, not run on hardware
//
#include <hip/hip_runtime.h>
#include <hip/hip_bf16.h>
#include <math.h>
#include <stdint.h>

// Problem constants (from reference)
#define BB    128
#define NN    512
#define INW   512
#define HID   512
#define RANK  64
#define POOL  8
// Derived
#define PRCOLS  (3 * POOL * RANK)   // 1536  (g,p,r) bilinear columns per side
#define LINCOLS (3 * HID)           // 1536  (g,h) linear columns
#define TOTCOLS (PRCOLS + PRCOLS + LINCOLS) // 4608: [LW | RW | LIN]
#define ACOLS   1024                // Acat row: [emb(512) | ph(512)]

#define KCHUNK      32              // K per WMMA issue (bf16)
#define BCOL_STRIDE 40              // padded LDS elems per column (32 data + 8 pad)

typedef __attribute__((ext_vector_type(16))) __bf16 v16bf;
typedef __attribute__((ext_vector_type(8)))  __bf16 v8bf;
typedef __attribute__((ext_vector_type(8)))  float  v8f;
typedef __attribute__((ext_vector_type(4)))  int    v4i;

#define GLOBAL_AS __attribute__((address_space(1)))
#define LDS_AS    __attribute__((address_space(3)))

// Exactly-typed pointers for the async builtin (revealed prototype: v4i AS1*, v4i AS3*)
#define ASYNC_GPTR(p) ((GLOBAL_AS v4i*)(uintptr_t)(const void*)(p))
#define ASYNC_LPTR(p) ((LDS_AS v4i*)(p))

#if defined(__has_builtin)
#if __has_builtin(__builtin_amdgcn_global_load_async_to_lds_b128) && \
    __has_builtin(__builtin_amdgcn_s_wait_asynccnt)
#define USE_ASYNC_LDS 1
#endif
#endif
#ifndef USE_ASYNC_LDS
#define USE_ASYNC_LDS 0
#endif

// ---------------------------------------------------------------------------
// One-time: convert weights to bf16 packs (output-column major, K contiguous)
// ---------------------------------------------------------------------------
__global__ void pack_weights_kernel(const float* __restrict__ wl,
                                    const float* __restrict__ wr,
                                    const float* __restrict__ ul,
                                    const float* __restrict__ ur,
                                    __bf16* __restrict__ WLp,
                                    __bf16* __restrict__ WRp,
                                    __bf16* __restrict__ ULURp) {
    int idx = blockIdx.x * blockDim.x + threadIdx.x;
    const int nPR = PRCOLS * INW;          // 786432
    const int nLIN = LINCOLS * ACOLS;      // 1572864
    if (idx < nPR) {
        WLp[idx] = (__bf16)wl[idx];
        WRp[idx] = (__bf16)wr[idx];
    }
    if (idx < nLIN) {
        int k = idx & (ACOLS - 1);
        int c = idx >> 10;                 // (g,h)
        float v = (k < INW) ? ul[c * INW + k] : ur[c * HID + (k - INW)];
        ULURp[idx] = (__bf16)v;
    }
}

// Zero node-0 rows of c-buffer and bf16 h-buffer (only unwritten state ever read)
__global__ void init_state_kernel(float* __restrict__ cbuf, __bf16* __restrict__ hbf) {
    int b = blockIdx.x;
    for (int h = threadIdx.x; h < HID; h += blockDim.x) {
        cbuf[(size_t)b * NN * HID + h] = 0.0f;
        hbf[(size_t)b * NN * HID + h] = (__bf16)0.0f;
    }
}

// ---------------------------------------------------------------------------
// Per step: Acat[b][0..511] = emb_t (cvt), [512..1023] = h_bf16[parent] gather
// ---------------------------------------------------------------------------
__global__ void build_A_kernel(const float* __restrict__ emb,
                               const int* __restrict__ conn,
                               const __bf16* __restrict__ hbf,
                               __bf16* __restrict__ Acat, int t) {
    int b = blockIdx.x;
    int par = conn[b * NN + t];
    const float* esrc = emb + ((size_t)b * NN + t) * INW;
    const __bf16* hsrc = hbf + ((size_t)b * NN + par) * HID;
    __bf16* dst = Acat + (size_t)b * ACOLS;
    for (int i = threadIdx.x; i < INW; i += blockDim.x) {
        dst[i] = (__bf16)esrc[i];
        dst[INW + i] = hsrc[i];
    }
}

// ---------------------------------------------------------------------------
// Per step block-structured GEMM:
//   G[128][4608] = [ emb_t x WLp | ph x WRp | [emb_t|ph] x ULURp ]
// 8 waves per WG; wave w owns rows 16w..16w+15; WG owns a 64-column slab.
// B chunks staged once per WG into double-buffered LDS (async when available),
// shared by all 8 waves -> 8x less B fetch traffic.
// ---------------------------------------------------------------------------
__device__ __forceinline__ v16bf load_frag(const __bf16* p) {
    v8bf lo = *(const v8bf*)p;        // K = base .. base+7
    v8bf hi = *(const v8bf*)(p + 16); // K = base+16 .. base+23
    return __builtin_shufflevector(lo, hi, 0, 1, 2, 3, 4, 5, 6, 7,
                                           8, 9, 10, 11, 12, 13, 14, 15);
}

__global__ void step_gemm_kernel(const __bf16* __restrict__ Acat,
                                 const __bf16* __restrict__ WLp,
                                 const __bf16* __restrict__ WRp,
                                 const __bf16* __restrict__ ULURp,
                                 float* __restrict__ G) {
    __shared__ __bf16 Bs[2][64 * BCOL_STRIDE];

    const int tid  = threadIdx.x;
    const int slab = blockIdx.x;          // 0..71
    const int wave = tid >> 5;            // 0..7 -> row tile
    const int lane = tid & 31;
    const int lo16 = lane & 15;
    const int half = lane >> 4;           // K sub-half per documented layout

    const int blockid   = slab / 24;      // 0=LW, 1=RW, 2=LIN
    const int slabInBlk = slab % 24;

    const __bf16* Bw;
    int koff, ksz, gcolbase;
    if (blockid == 0)      { Bw = WLp;   koff = 0;   ksz = INW;   gcolbase = slabInBlk * 64; }
    else if (blockid == 1) { Bw = WRp;   koff = INW; ksz = HID;   gcolbase = PRCOLS + slabInBlk * 64; }
    else                   { Bw = ULURp; koff = 0;   ksz = ACOLS; gcolbase = 2 * PRCOLS + slabInBlk * 64; }

    const int cbase = slabInBlk * 64;
    const int row   = wave * 16 + lo16;
    const __bf16* arow = Acat + (size_t)row * ACOLS + koff + half * 8;

    // B staging: thread t copies 16B piece (t&3) of local column (t>>2)
    const int ccol_loc = tid >> 2;
    const int piece    = tid & 3;
    const __bf16* gsrc = Bw + (size_t)(cbase + ccol_loc) * ksz + piece * 8;
    __bf16* l0 = &Bs[0][ccol_loc * BCOL_STRIDE + piece * 8];
    __bf16* l1 = &Bs[1][ccol_loc * BCOL_STRIDE + piece * 8];

    v8f acc0 = {}, acc1 = {}, acc2 = {}, acc3 = {};

#if USE_ASYNC_LDS
    __builtin_amdgcn_global_load_async_to_lds_b128(ASYNC_GPTR(gsrc), ASYNC_LPTR(l0), 0, 0);
#else
    *(v8bf*)l0 = *(const v8bf*)gsrc;
#endif

    int s = 0;
    for (int kc = 0; kc < ksz; kc += KCHUNK, s ^= 1) {
        // Prefetch next chunk into the other buffer; wait for current chunk.
        if (kc + KCHUNK < ksz) {
            __bf16* lnxt = (s == 0) ? l1 : l0;
#if USE_ASYNC_LDS
            __builtin_amdgcn_global_load_async_to_lds_b128(
                ASYNC_GPTR(gsrc + kc + KCHUNK), ASYNC_LPTR(lnxt), 0, 0);
            __builtin_amdgcn_s_wait_asynccnt(1);  // in-flight prefetch may stay out
#else
            *(v8bf*)lnxt = *(const v8bf*)(gsrc + kc + KCHUNK);
#endif
        } else {
#if USE_ASYNC_LDS
            __builtin_amdgcn_s_wait_asynccnt(0);
#endif
        }
        __syncthreads();

        v16bf a = load_frag(arow + kc);
        const __bf16* bb = &Bs[s][0];
        v16bf b0 = load_frag(bb + (0 * 16 + lo16) * BCOL_STRIDE + half * 8);
        v16bf b1 = load_frag(bb + (1 * 16 + lo16) * BCOL_STRIDE + half * 8);
        v16bf b2 = load_frag(bb + (2 * 16 + lo16) * BCOL_STRIDE + half * 8);
        v16bf b3 = load_frag(bb + (3 * 16 + lo16) * BCOL_STRIDE + half * 8);

        acc0 = __builtin_amdgcn_wmma_f32_16x16x32_bf16(false, a, false, b0, (short)0, acc0, false, false);
        acc1 = __builtin_amdgcn_wmma_f32_16x16x32_bf16(false, a, false, b1, (short)0, acc1, false, false);
        acc2 = __builtin_amdgcn_wmma_f32_16x16x32_bf16(false, a, false, b2, (short)0, acc2, false, false);
        acc3 = __builtin_amdgcn_wmma_f32_16x16x32_bf16(false, a, false, b3, (short)0, acc3, false, false);

        __syncthreads();  // protect buffer s before it is refilled (next iter)
    }

    // C/D layout: lane<16 -> N=lane, M=v; lane>=16 -> N=lane-16, M=v+8
    #pragma unroll
    for (int v = 0; v < 8; ++v) {
        int r = wave * 16 + half * 8 + v;
        float* gr = G + (size_t)r * TOTCOLS + gcolbase + lo16;
        gr[0 * 16] = acc0[v];
        gr[1 * 16] = acc1[v];
        gr[2 * 16] = acc2[v];
        gr[3 * 16] = acc3[v];
    }
}

// ---------------------------------------------------------------------------
// Per step pointwise: pooled rank reduction, wo projection, gates, c/h update.
// ---------------------------------------------------------------------------
__global__ void step_pointwise_kernel(const float* __restrict__ G,
                                      const float* __restrict__ wo,
                                      const float* __restrict__ bias,
                                      const int* __restrict__ conn,
                                      float* __restrict__ cbuf,
                                      __bf16* __restrict__ hbf,
                                      float* __restrict__ out, int t) {
    __shared__ float pooled[3 * POOL];
    __shared__ float gate_s[3 * HID];

    const int b = blockIdx.x;
    const int tid = threadIdx.x;
    const float* Gr = G + (size_t)b * TOTCOLS;

    if (tid < 3 * POOL) {
        int base = tid * RANK;
        float ssum = 0.0f;
        #pragma unroll 8
        for (int r = 0; r < RANK; ++r)
            ssum += Gr[base + r] * Gr[PRCOLS + base + r];
        pooled[tid] = ssum;
    }
    __syncthreads();

    for (int c = tid; c < 3 * HID; c += blockDim.x) {
        int g = c / HID;
        float v = Gr[2 * PRCOLS + c] + bias[c];
        const float* wop = wo + (size_t)c * POOL;
        #pragma unroll
        for (int p = 0; p < POOL; ++p)
            v += pooled[g * POOL + p] * wop[p];
        gate_s[c] = v;
    }
    __syncthreads();

    const int par = conn[b * NN + t];
    for (int h = tid; h < HID; h += blockDim.x) {
        float f = 1.0f / (1.0f + expf(-gate_s[h]));
        float o = 1.0f / (1.0f + expf(-gate_s[HID + h]));
        float z = tanhf(gate_s[2 * HID + h]);
        float pc = cbuf[((size_t)b * NN + par) * HID + h];
        float c = pc * f + (1.0f - f) * z;
        float hv = o * tanhf(c);
        cbuf[((size_t)b * NN + t) * HID + h] = c;
        out[((size_t)b * NN + t) * HID + h] = hv;
        hbf[((size_t)b * NN + t) * HID + h] = (__bf16)hv;
    }
}

// ---------------------------------------------------------------------------
extern "C" void kernel_launch(void* const* d_in, const int* in_sizes, int n_in,
                              void* d_out, int out_size, void* d_ws, size_t ws_size,
                              hipStream_t stream) {
    const float* emb  = (const float*)d_in[0];   // (B,N,IN)
    const int*   conn = (const int*)d_in[1];     // (B,N)
    // d_in[2] node_mask: unused by the reference math
    const float* wl   = (const float*)d_in[3];   // (3,POOL,RANK,IN)
    const float* wr   = (const float*)d_in[4];   // (3,POOL,RANK,HID)
    const float* wo   = (const float*)d_in[5];   // (3,HID,POOL)
    const float* ul   = (const float*)d_in[6];   // (3,HID,IN)
    const float* ur   = (const float*)d_in[7];   // (3,HID,HID)
    const float* bias = (const float*)d_in[8];   // (3,HID)
    float* out = (float*)d_out;                  // (B,N,HID) f32

    char* ws = (char*)d_ws;
    size_t off = 0;
    auto carve = [&](size_t bytes) -> void* {
        void* p = ws + off;
        off = (off + bytes + 255) & ~(size_t)255;
        return p;
    };
    __bf16* WLp   = (__bf16*)carve((size_t)PRCOLS * INW * sizeof(__bf16));
    __bf16* WRp   = (__bf16*)carve((size_t)PRCOLS * HID * sizeof(__bf16));
    __bf16* ULURp = (__bf16*)carve((size_t)LINCOLS * ACOLS * sizeof(__bf16));
    __bf16* Acat  = (__bf16*)carve((size_t)BB * ACOLS * sizeof(__bf16));
    float*  G     = (float*)carve((size_t)BB * TOTCOLS * sizeof(float));
    float*  cbuf  = (float*)carve((size_t)BB * NN * HID * sizeof(float));
    __bf16* hbf   = (__bf16*)carve((size_t)BB * NN * HID * sizeof(__bf16));
    (void)ws_size; (void)in_sizes; (void)n_in; (void)out_size;

    {
        int total = LINCOLS * ACOLS;     // largest pack
        pack_weights_kernel<<<(total + 255) / 256, 256, 0, stream>>>(
            wl, wr, ul, ur, WLp, WRp, ULURp);
        init_state_kernel<<<BB, 256, 0, stream>>>(cbuf, hbf);
    }

    for (int t = 0; t < NN; ++t) {
        build_A_kernel<<<BB, 256, 0, stream>>>(emb, conn, hbf, Acat, t);
        step_gemm_kernel<<<72, 256, 0, stream>>>(Acat, WLp, WRp, ULURp, G);
        step_pointwise_kernel<<<BB, 256, 0, stream>>>(G, wo, bias, conn,
                                                      cbuf, hbf, out, t);
    }
}
